// ObjectPermanenceModule_85830626443554
// MI455X (gfx1250) — matrix operand order repaired and checked by reading, not compiled, not run on hardware
//
#include <hip/hip_runtime.h>
#include <hip/hip_bf16.h>

// ---- CDNA5 (gfx1250) types for V_WMMA_F32_16X16X32_BF16 ----
typedef __bf16 bf16t;
typedef __attribute__((ext_vector_type(16))) __bf16 v16bf;
typedef __attribute__((ext_vector_type(8)))  float  v8f;

#define B_   16
#define N_   10
#define F_   512
#define H_   256
#define GHW  32
#define ROWS 160      // B*N
#define KCAT 1536     // [cur|prev|vel]

__device__ __forceinline__ float sig_(float v) { return 1.0f / (1.0f + __expf(-v)); }

// ---------------------------------------------------------------------------
// Stage A (per batch b): sim argmax over the 10 slots at grid cell (0,0),
// cur[b] = 0.7*x + 0.3*grid_slot, then occupancy-logits argmax (softmax-free).
// ---------------------------------------------------------------------------
__global__ void k_stageA(const float* __restrict__ x, const float* __restrict__ gs,
                         const float* __restrict__ Wmap, const float* __restrict__ bmap,
                         float* __restrict__ cur, int* __restrict__ cellidx)
{
    __shared__ float xs[F_];
    __shared__ float curs[F_];
    __shared__ float red[256];
    __shared__ int   redi[256];
    __shared__ float sim[N_];
    __shared__ int   ks_s;
    const int b = blockIdx.x, t = threadIdx.x;

    for (int f = t; f < F_; f += 256) xs[f] = x[b * F_ + f];
    __syncthreads();

    const float* cell0 = gs + (size_t)b * GHW * GHW * N_ * F_;   // gh=0, gw=0
    for (int k = 0; k < N_; ++k) {
        float p = 0.f;
        const float* gk = cell0 + (size_t)k * F_;
        for (int f = t; f < F_; f += 256) p += xs[f] * gk[f];
        red[t] = p; __syncthreads();
        for (int s = 128; s > 0; s >>= 1) { if (t < s) red[t] += red[t + s]; __syncthreads(); }
        if (t == 0) sim[k] = red[0];
        __syncthreads();
    }
    if (t == 0) {                       // first-occurrence argmax (jnp semantics)
        int best = 0; float bv = sim[0];
        for (int k = 1; k < N_; ++k) if (sim[k] > bv) { bv = sim[k]; best = k; }
        ks_s = best;
    }
    __syncthreads();

    const float* gk = cell0 + (size_t)ks_s * F_;
    for (int f = t; f < F_; f += 256) {
        float c = 0.7f * xs[f] + 0.3f * gk[f];
        curs[f] = c; cur[b * F_ + f] = c;
    }
    __syncthreads();

    // occ_logits = cur @ Wmap + bmap ; argmax over 1024 (softmax is monotone)
    float bv = -3.4e38f; int bi = 0;
    for (int c = t; c < 1024; c += 256) {
        float a = bmap[c];
        for (int f = 0; f < F_; ++f) a += curs[f] * Wmap[(size_t)f * 1024 + c];
        if (a > bv) { bv = a; bi = c; }
    }
    red[t] = bv; redi[t] = bi; __syncthreads();
    for (int s = 128; s > 0; s >>= 1) {
        if (t < s) {
            if (red[t + s] > red[t] || (red[t + s] == red[t] && redi[t + s] < redi[t])) {
                red[t] = red[t + s]; redi[t] = redi[t + s];
            }
        }
        __syncthreads();
    }
    if (t == 0) cellidx[b] = redi[0];
}

// ---------------------------------------------------------------------------
// Pack a row-major f32 weight [K, Nout] into bf16 B-fragment-native order:
// Bp[((k/32)*Nout + n)*32 + k%32], so a column's 32 K-values per k-tile are contiguous.
// ---------------------------------------------------------------------------
__global__ void k_pack(const float* __restrict__ W, bf16t* __restrict__ out, int K, int Nout)
{
    int idx = blockIdx.x * blockDim.x + threadIdx.x;
    if (idx >= K * Nout) return;
    int k = idx / Nout, n = idx - k * Nout;
    out[((size_t)(k >> 5) * Nout + n) * 32 + (k & 31)] = (bf16t)W[idx];
}

// ---------------------------------------------------------------------------
// Xcat[r=b*10+n, 0:1536] = [cur[b] | prev[b,n] | x[b]-prev[b,n]]  (bf16)
// ---------------------------------------------------------------------------
__global__ void k_build(const float* __restrict__ x, const float* __restrict__ prev,
                        const float* __restrict__ cur, bf16t* __restrict__ Xcat)
{
    const int r = blockIdx.x, t = threadIdx.x;
    const int b = r / N_;
    const float* pv = prev + (size_t)r * F_;
    const float* cb = cur  + (size_t)b * F_;
    const float* xb = x    + (size_t)b * F_;
    bf16t* row = Xcat + (size_t)r * KCAT;
    for (int c = t; c < KCAT; c += 256) {
        float v;
        if (c < F_)            v = cb[c];
        else if (c < 2 * F_)   v = pv[c - F_];
        else                   v = xb[c - 2 * F_] - pv[c - 2 * F_];
        row[c] = (bf16t)v;
    }
}

// ---------------------------------------------------------------------------
// Generic bf16 WMMA GEMM: C[160,Nout] = epi(A[160,K] @ B[K,Nout] + bias).
// One wave per 16x16 tile. A row-major (lda), B pre-packed by k_pack.
// Optional rank-1 occlusion column (Wg1 row 1536), relu, f32/bf16 stores.
// ---------------------------------------------------------------------------
__global__ void k_gemm(const bf16t* __restrict__ A, int lda, int Ktiles,
                       const bf16t* __restrict__ Bp, int Nout, int Ntiles,
                       const float* __restrict__ bias,
                       const float* __restrict__ occW, const float* __restrict__ occl,
                       float* __restrict__ outF, bf16t* __restrict__ outB, int relu)
{
    const int wid  = blockIdx.x * (blockDim.x >> 5) + (threadIdx.x >> 5);
    const int lane = threadIdx.x & 31;
    const int tiles = 10 * Ntiles;                 // Mtiles = 160/16 = 10
    if (wid >= tiles) return;                      // whole-wave guard (EXEC all-1 inside)
    const int tm = wid / Ntiles, tn = wid - tm * Ntiles;
    const int hf = lane >> 4;                      // lane half (ISA 7.12.2 layouts)
    const int ln = lane & 15;
    const int n = tn * 16 + ln;
    const int mbase = tm * 16 + hf * 8;

    const float bn = bias[n];
    v8f acc;
    #pragma unroll
    for (int i = 0; i < 8; ++i) acc[i] = bn;

    // A frag: lanes0-15 row m=ln, K = {0..7}∪{16..23}; lanes16-31 K = {8..15}∪{24..31}
    const bf16t* arow = A + (size_t)(tm * 16 + ln) * lda + hf * 8;
    // B frag: lanes0-15 K = {0..7}∪{16..23} at col n; lanes16-31 K = {8..15}∪{24..31}
    const bf16t* bcol = Bp + (size_t)n * 32 + hf * 8;

    for (int kt = 0; kt < Ktiles; ++kt) {
        union { uint4 u[2]; v16bf v; } ua, ub;
        const bf16t* ap = arow + kt * 32;
        ua.u[0] = *(const uint4*)(ap);             // K base+0..7
        ua.u[1] = *(const uint4*)(ap + 16);        // K base+16..23
        const bf16t* bp = bcol + (size_t)kt * Nout * 32;
        ub.u[0] = *(const uint4*)(bp);             // K base+0..7
        ub.u[1] = *(const uint4*)(bp + 16);        // K base+16..23
        acc = __builtin_amdgcn_wmma_f32_16x16x32_bf16(
                  false, ua.v, false, ub.v, (short)0, acc, false, false);
    }

    if (occW) {                                    // + occlusion[r] * Wg1[1536, n]
        const float ow = occW[n];
        #pragma unroll
        for (int i = 0; i < 8; ++i) acc[i] += occl[mbase + i] * ow;
    }
    if (relu) {
        #pragma unroll
        for (int i = 0; i < 8; ++i) acc[i] = acc[i] > 0.f ? acc[i] : 0.f;
    }
    if (outF) {
        #pragma unroll
        for (int i = 0; i < 8; ++i) outF[(size_t)(mbase + i) * Nout + n] = acc[i];
    }
    if (outB) {
        #pragma unroll
        for (int i = 0; i < 8; ++i) outB[(size_t)(mbase + i) * Nout + n] = (bf16t)acc[i];
    }
}

// out[r] = sigmoid(dot(Hin[r,:cols], w) + b)   (tiny; 160 rows)
__global__ void k_rowdot(const float* __restrict__ Hin, int cols,
                         const float* __restrict__ w, const float* __restrict__ bsc,
                         float* __restrict__ out, int rows)
{
    int t = blockIdx.x * blockDim.x + threadIdx.x;
    if (t >= rows) return;
    const float* h = Hin + (size_t)t * cols;
    float a = bsc[0];
    for (int c = 0; c < cols; ++c) a += h[c] * w[c];
    out[t] = sig_(a);
}

// occluded[n] = mean_b(occlusion[b,n]) > 0.5
__global__ void k_occluded(const float* __restrict__ occl, float* __restrict__ occd)
{
    int t = threadIdx.x;
    if (t < N_) {
        float s = 0.f;
        for (int b = 0; b < B_; ++b) s += occl[b * N_ + t];
        occd[t] = (s * (1.0f / B_) > 0.5f) ? 1.f : 0.f;
    }
}

// fused = pg*cur + (1-pg)*(occluded ? 0.5*pred + 0.5*grid_feat : pred)
__global__ void k_final(const float* __restrict__ cur, const float* __restrict__ pred,
                        const float* __restrict__ gs, const int* __restrict__ cellidx,
                        const float* __restrict__ occd, const float* __restrict__ pg,
                        float* __restrict__ out)
{
    const int r = blockIdx.x, t = threadIdx.x;
    const int b = r / N_, n = r - b * N_;
    const float g = pg[r];
    const bool  od = occd[n] > 0.5f;
    const int cell = cellidx[b];
    const int gh = cell / GHW, gw = cell - gh * GHW;
    const float* gf = gs + ((((size_t)b * GHW + gh) * GHW + gw) * N_ + n) * F_;
    const float* cb = cur + (size_t)b * F_;
    const float* pr = pred + (size_t)r * F_;
    float* o = out + (size_t)r * F_;
    for (int f = t; f < F_; f += 256) {
        float p = pr[f];
        if (od) p = 0.5f * p + 0.5f * gf[f];
        o[f] = g * cb[f] + (1.f - g) * p;
    }
}

extern "C" void kernel_launch(void* const* d_in, const int* in_sizes, int n_in,
                              void* d_out, int out_size, void* d_ws, size_t ws_size,
                              hipStream_t stream)
{
    const float* x     = (const float*)d_in[0];
    const float* prev  = (const float*)d_in[1];
    const float* gs    = (const float*)d_in[2];
    const float* Wocc1 = (const float*)d_in[3];
    const float* bocc1 = (const float*)d_in[4];
    const float* Wocc2 = (const float*)d_in[5];
    const float* bocc2 = (const float*)d_in[6];
    const float* Wp1   = (const float*)d_in[7];
    const float* bp1   = (const float*)d_in[8];
    const float* Wp2   = (const float*)d_in[9];
    const float* bp2   = (const float*)d_in[10];
    // d_in[11]=Wpc, d_in[12]=bpc : outputs deleted in reference -> unused
    const float* Wg1   = (const float*)d_in[13];
    const float* bg1   = (const float*)d_in[14];
    const float* Wgp   = (const float*)d_in[15];
    const float* bgp   = (const float*)d_in[16];
    // d_in[17]=Wga, d_in[18]=bga : deleted -> unused
    const float* Wmap  = (const float*)d_in[19];
    const float* bmap  = (const float*)d_in[20];

    size_t off = 0;
    char* base = (char*)d_ws;
    auto carve = [&](size_t bytes) -> void* {
        void* p = base + off;
        off += (bytes + 255) & ~(size_t)255;
        return p;
    };
    float* curB    = (float*)carve((size_t)B_ * F_ * 4);        // cur [16,512]
    int*   cellidx = (int*)  carve(B_ * 4);
    bf16t* Xcat    = (bf16t*)carve((size_t)ROWS * KCAT * 2);    // [160,1536] bf16
    bf16t* Wocc1p  = (bf16t*)carve((size_t)1024 * 256 * 2);
    bf16t* Wp1p    = (bf16t*)carve((size_t)1536 * 256 * 2);
    bf16t* Wp2p    = (bf16t*)carve((size_t)256 * 512 * 2);
    bf16t* Wg1p    = (bf16t*)carve((size_t)1536 * 256 * 2);
    float* hocc    = (float*)carve((size_t)ROWS * H_ * 4);
    bf16t* hp      = (bf16t*)carve((size_t)ROWS * H_ * 2);
    float* hg      = (float*)carve((size_t)ROWS * H_ * 4);
    float* pred    = (float*)carve((size_t)ROWS * F_ * 4);
    float* occl    = (float*)carve(ROWS * 4);
    float* occd    = (float*)carve(N_ * 4);
    float* pg      = (float*)carve(ROWS * 4);
    (void)ws_size; (void)in_sizes; (void)n_in; (void)out_size;

    // Stage A: cur + occupancy cell index (per batch)
    k_stageA<<<B_, 256, 0, stream>>>(x, gs, Wmap, bmap, curB, cellidx);

    // Weight packing (f32 -> bf16, fragment-native order)
    k_pack<<<(1024 * 256) / 256, 256, 0, stream>>>(Wocc1, Wocc1p, 1024, 256);
    k_pack<<<(1536 * 256) / 256, 256, 0, stream>>>(Wp1,   Wp1p,   1536, 256);
    k_pack<<<(256  * 512) / 256, 256, 0, stream>>>(Wp2,   Wp2p,   256,  512);
    k_pack<<<(1536 * 256) / 256, 256, 0, stream>>>(Wg1,   Wg1p,   1536, 256);

    // Activation matrix [cur|prev|vel]
    k_build<<<ROWS, 256, 0, stream>>>(x, prev, curB, Xcat);

    // h_occ = relu(Xcat[:, :1024] @ Wocc1 + bocc1)       (K=1024 -> 32 ktiles)
    k_gemm<<<20, 256, 0, stream>>>(Xcat, KCAT, 32, Wocc1p, 256, 16, bocc1,
                                   nullptr, nullptr, hocc, nullptr, 1);
    // occlusion = sigmoid(h_occ @ Wocc2 + bocc2); occluded[n] = mean_b > 0.5
    k_rowdot<<<1, 256, 0, stream>>>(hocc, 256, Wocc2, bocc2, occl, ROWS);
    k_occluded<<<1, 32, 0, stream>>>(occl, occd);

    // h_p = relu(Xcat @ Wp1 + bp1)  (bf16 out);  predicted = h_p @ Wp2 + bp2
    k_gemm<<<20, 256, 0, stream>>>(Xcat, KCAT, 48, Wp1p, 256, 16, bp1,
                                   nullptr, nullptr, nullptr, hp, 1);
    k_gemm<<<40, 256, 0, stream>>>(hp, 256, 8, Wp2p, 512, 32, bp2,
                                   nullptr, nullptr, pred, nullptr, 0);

    // h_g = relu(Xcat @ Wg1[:1536] + occl*Wg1[1536] + bg1); pos_gate = sigmoid(h_g @ Wgp + bgp)
    k_gemm<<<20, 256, 0, stream>>>(Xcat, KCAT, 48, Wg1p, 256, 16, bg1,
                                   Wg1 + (size_t)1536 * 256, occl, hg, nullptr, 1);
    k_rowdot<<<1, 256, 0, stream>>>(hg, 256, Wgp, bgp, pg, ROWS);

    // Fusion + occluded grid blend
    k_final<<<ROWS, 256, 0, stream>>>(curB, pred, gs, cellidx, occd, pg, (float*)d_out);
}
